// HANEncoder_3745211482559
// MI455X (gfx1250) — matrix operand order repaired and checked by reading, not compiled, not run on hardware
//
#include <hip/hip_runtime.h>
#include <hip/hip_bf16.h>

typedef _Float16 half_t;
typedef __attribute__((ext_vector_type(16))) _Float16 v16h;
typedef __attribute__((ext_vector_type(8)))  _Float16 v8h;
typedef __attribute__((ext_vector_type(8)))  float    v8f;

#define NC_   20000
#define NQ_   100000
#define ECQ_  800000
#define EQC_  800000
#define ECC_  200000
#define CH_   128
#define HD_   8
#define HID_  768
#define SLOPE_ 0.2f
#define MT_   4      // M tiles per wave in the GEMM (B fragments stay register-resident)

// ---------- order-preserving float <-> uint for atomic max (0x0 == -inf sentinel-safe) ----------
__device__ __forceinline__ unsigned fenc(float f){
  unsigned u = __float_as_uint(f);
  return (u & 0x80000000u) ? ~u : (u | 0x80000000u);
}
__device__ __forceinline__ float fdec(unsigned u){
  return __uint_as_float((u & 0x80000000u) ? (u & 0x7FFFFFFFu) : ~u);
}

// ---------- f32 -> f16 convert (row-major, for A-side activations) ----------
__global__ void cvt16_kernel(const float* __restrict__ in, half_t* __restrict__ out, int n){
  int t = blockIdx.x * blockDim.x + threadIdx.x;
  if (t < n) out[t] = (half_t)in[t];
}

// ---------- f32 -> f16 convert + transpose: in[K=128,N] -> out[N,128] (weights) ----------
__global__ void cvtT_kernel(const float* __restrict__ in, half_t* __restrict__ out, int N){
  int t = blockIdx.x * blockDim.x + threadIdx.x;
  if (t >= 128 * N) return;
  int k = t / N, n = t - k * N;            // read coalesced over in
  out[(size_t)n * 128 + k] = (half_t)in[t];
}

// ---------- WMMA GEMM: C[M,N] = A[M,128] * B[128,N] (+bias); Bt is pre-transposed [N,128] ----------
// mode 0: store C+bias (f32). mode 1: colsum[n] += sum_m tanh(C+bias) (semantic path, no store).
__global__ void __launch_bounds__(128)
wmma_gemm128(const half_t* __restrict__ A, const half_t* __restrict__ Bt,
             const float* __restrict__ bias, float* __restrict__ out,
             float* __restrict__ colsum, int M, int N, int mode){
  const int lane = threadIdx.x;
  const int nt   = blockIdx.x;
  const int wave = blockIdx.y * blockDim.y + threadIdx.y;
  const int mt0  = wave * MT_;
  if (mt0 * 16 >= M) return;               // wave-uniform exit: EXEC all-1s for WMMA
  const int colL = lane & 15;
  const int hsel = lane >> 4;              // half-wave select
  const int ncol = nt * 16 + colL;

  // B fragments: element i = B[kk*32 + hsel*16 + i, ncol] = Bt[ncol*128 + kk*32 + hsel*16 + i]
  // -> 32 contiguous bytes per fragment (vector loads, no scalar gather)
  const half_t* bp = Bt + (size_t)ncol * 128 + hsel * 16;
  v16h bfr[4];
  #pragma unroll
  for (int kk = 0; kk < 4; ++kk)
    bfr[kk] = *(const v16h*)(bp + kk * 32);

  float csum = 0.0f;
  const float bv = bias ? bias[ncol] : 0.0f;

  #pragma unroll
  for (int m = 0; m < MT_; ++m){
    const int mt = mt0 + m;
    if (mt * 16 >= M) break;               // wave-uniform (same wave id for all lanes)
    // Hoist all A loads for this tile: 8 b128 loads issue as one clause, then the
    // loadcnt waiter counts down across the WMMA chain instead of full-waiting per step.
    const half_t* ap = A + (size_t)(mt * 16 + colL) * CH_;
    v8h lo[4], hi[4];
    #pragma unroll
    for (int kk = 0; kk < 4; ++kk){
      const int kb = kk * 32 + hsel * 16;
      lo[kk] = *(const v8h*)(ap + kb);
      hi[kk] = *(const v8h*)(ap + kb + 8);
    }
    v8f acc = {};
    #pragma unroll
    for (int kk = 0; kk < 4; ++kk){
      v16h af;
      #pragma unroll
      for (int i = 0; i < 8; ++i){ af[i] = lo[kk][i]; af[i + 8] = hi[kk][i]; }
      acc = __builtin_amdgcn_wmma_f32_16x16x32_f16(false, af, false, bfr[kk],
                                                   (short)0, acc, false, false);
    }
    const int mbase = mt * 16 + hsel * 8;  // C layout: vgpr r -> row mbase+r, col = ncol
    if (mode == 0){
      #pragma unroll
      for (int r = 0; r < 8; ++r)
        out[(size_t)(mbase + r) * N + ncol] = acc[r] + bv;
    } else {
      #pragma unroll
      for (int r = 0; r < 8; ++r) csum += tanhf(acc[r] + bv);
    }
  }
  if (mode != 0) atomicAdd(&colsum[ncol], csum);
}

// ---------- per-node attention scores: a[n,h] = sum_d h[n,h*16+d]*att[h,d] ----------
__global__ void att_score_kernel(const float* __restrict__ h, const float* __restrict__ att,
                                 float* __restrict__ out, int n_nodes){
  int t = blockIdx.x * blockDim.x + threadIdx.x;
  if (t >= n_nodes * HD_) return;
  int n = t >> 3, hd = t & 7;
  const float* hp = h + (size_t)n * CH_ + hd * 16;
  const float* ap = att + hd * 16;
  float s = 0.f;
  #pragma unroll
  for (int i = 0; i < 16; ++i) s += hp[i] * ap[i];
  out[t] = s;
}

// ---------- pass A: per-edge alpha, atomic segment max per (dst,head) ----------
__global__ void edge_max_kernel(const int* __restrict__ row, const int* __restrict__ col,
                                const float* __restrict__ asrc, const float* __restrict__ adst,
                                unsigned* __restrict__ umax, int E){
  int t = blockIdx.x * blockDim.x + threadIdx.x;
  if (t >= E * HD_) return;
  int e = t >> 3, hd = t & 7;
  int r = row[e], c = col[e];
  float a = asrc[r * HD_ + hd] + adst[c * HD_ + hd];
  a = a > 0.f ? a : SLOPE_ * a;
  atomicMax(&umax[c * HD_ + hd], fenc(a));
}

// ---------- pass B: wave32 per edge; e = exp(alpha-max); denom += e; agg += h_src * e ----------
__global__ void __launch_bounds__(256)
edge_scatter_kernel(const int* __restrict__ row, const int* __restrict__ col,
                    const float* __restrict__ asrc, const float* __restrict__ adst,
                    const unsigned* __restrict__ umax, const float* __restrict__ hsrc,
                    float* __restrict__ denom, float* __restrict__ agg, int E){
  const int lane = threadIdx.x;
  const int e = blockIdx.x * blockDim.y + threadIdx.y;
  if (e >= E) return;
  const int r = row[e], c = col[e];
  float eh = 0.f;
  if (lane < HD_){
    float a = asrc[r * HD_ + lane] + adst[c * HD_ + lane];
    a = a > 0.f ? a : SLOPE_ * a;
    eh = expf(a - fdec(umax[c * HD_ + lane]));
    atomicAdd(&denom[c * HD_ + lane], eh);
  }
  const int b = lane >> 4;                 // channel lane+32j -> head 2j+b
  float e0 = __shfl(eh, b,     32);
  float e1 = __shfl(eh, b + 2, 32);
  float e2 = __shfl(eh, b + 4, 32);
  float e3 = __shfl(eh, b + 6, 32);
  const float* hp = hsrc + (size_t)r * CH_;
  float* apg = agg + (size_t)c * CH_;
  atomicAdd(apg + lane,      hp[lane]      * e0);
  atomicAdd(apg + lane + 32, hp[lane + 32] * e1);
  atomicAdd(apg + lane + 64, hp[lane + 64] * e2);
  atomicAdd(apg + lane + 96, hp[lane + 96] * e3);
}

// ---------- pass C: out = relu(agg / (denom + eps)) in place ----------
__global__ void finalize_kernel(float* __restrict__ agg, const float* __restrict__ denom, int n_nodes){
  int t = blockIdx.x * blockDim.x + threadIdx.x;
  if (t >= n_nodes * CH_) return;
  int n = t >> 7, d = t & 127;
  float v = agg[t] / (denom[n * HD_ + (d >> 4)] + 1e-16f);
  agg[t] = v > 0.f ? v : 0.f;
}

// ---------- semantic attention: scores from column sums, softmax over K<=2 ----------
__global__ void semantic_attn_kernel(const float* __restrict__ colsum, const float* __restrict__ qv,
                                     float* __restrict__ attn, int K, float invN){
  __shared__ float sh[128];
  __shared__ float sc[2];
  int t = threadIdx.x;
  for (int k = 0; k < K; ++k){
    sh[t] = qv[t] * colsum[k * 128 + t] * invN;
    __syncthreads();
    for (int s = 64; s > 0; s >>= 1){ if (t < s) sh[t] += sh[t + s]; __syncthreads(); }
    if (t == 0) sc[k] = sh[0];
    __syncthreads();
  }
  if (t == 0){
    float m = sc[0];
    for (int k = 1; k < K; ++k) m = fmaxf(m, sc[k]);
    float den = 0.f, ev[2];
    for (int k = 0; k < K; ++k){ ev[k] = expf(sc[k] - m); den += ev[k]; }
    for (int k = 0; k < K; ++k) attn[k] = ev[k] / den;
  }
}

// ---------- combine metapaths (+ optional exact GELU), emit f16 for next GEMM ----------
__global__ void combine_kernel(const float* __restrict__ x0, const float* __restrict__ x1,
                               const float* __restrict__ attn, int K, int total,
                               half_t* __restrict__ of16, int do_gelu){
  int t = blockIdx.x * blockDim.x + threadIdx.x;
  if (t >= total) return;
  float v = attn[0] * x0[t];
  if (K > 1) v += attn[1] * x1[t];
  if (do_gelu) v = 0.5f * v * (1.0f + erff(v * 0.70710678118654752f));
  of16[t] = (half_t)v;
}

// ==================== host side ====================
struct HanP {
  const float *att_dst_cc, *att_dst_cq, *att_dst_qc;
  const float *att_src_cc, *att_src_cq, *att_src_qc;
  const float *k_b, *k_w;
  const float *proj_c_b, *proj_c_w, *proj_q_b, *proj_q_w;
  const float *q;
};

static inline size_t al256(size_t x){ return (x + 255) & ~(size_t)255; }

extern "C" void kernel_launch(void* const* d_in, const int* in_sizes, int n_in,
                              void* d_out, int out_size, void* d_ws, size_t ws_size,
                              hipStream_t stream) {
  (void)in_sizes; (void)n_in; (void)out_size; (void)ws_size;

  const float* xc    = (const float*)d_in[0];
  const float* xq    = (const float*)d_in[1];
  const int*   ei_cq = (const int*)d_in[2];
  const int*   ei_qc = (const int*)d_in[3];
  const int*   ei_cc = (const int*)d_in[4];
  // params flattened: top-level sorted keys -> cproj_b, cproj_w, han1(13), han2(13), qproj_b, qproj_w
  const float* cproj_b = (const float*)d_in[5];
  const float* cproj_w = (const float*)d_in[6];
  auto grab_han = [&](int base)->HanP{
    HanP p;
    p.att_dst_cc = (const float*)d_in[base + 0];
    p.att_dst_cq = (const float*)d_in[base + 1];
    p.att_dst_qc = (const float*)d_in[base + 2];
    p.att_src_cc = (const float*)d_in[base + 3];
    p.att_src_cq = (const float*)d_in[base + 4];
    p.att_src_qc = (const float*)d_in[base + 5];
    p.k_b        = (const float*)d_in[base + 6];
    p.k_w        = (const float*)d_in[base + 7];
    p.proj_c_b   = (const float*)d_in[base + 8];
    p.proj_c_w   = (const float*)d_in[base + 9];
    p.proj_q_b   = (const float*)d_in[base + 10];
    p.proj_q_w   = (const float*)d_in[base + 11];
    p.q          = (const float*)d_in[base + 12];
    return p;
  };
  HanP h1 = grab_han(7);
  HanP h2 = grab_han(20);
  const float* qproj_b = (const float*)d_in[33];
  const float* qproj_w = (const float*)d_in[34];
  float* out_c = (float*)d_out;
  float* out_q = (float*)d_out + (size_t)NC_ * HID_;

  // ---- workspace carve ----
  char* base = (char*)d_ws; size_t off = 0;
  auto carve = [&](size_t bytes)->void*{ void* p = base + off; off = al256(off + bytes); return p; };
  half_t* xc16    = (half_t*)carve((size_t)NC_ * CH_ * 2);
  half_t* xq16    = (half_t*)carve((size_t)NQ_ * CH_ * 2);
  half_t* cc16    = (half_t*)carve((size_t)NC_ * CH_ * 2);   // f16 of agg_cc for semantic GEMM
  half_t* wbufA   = (half_t*)carve((size_t)CH_ * HID_ * 2);  // transposed f16 weights
  half_t* wbufB   = (half_t*)carve((size_t)CH_ * HID_ * 2);
  float*  h_c     = (float*)carve((size_t)NC_ * CH_ * 4);
  float*  h_q     = (float*)carve((size_t)NQ_ * CH_ * 4);
  float*  as_cq   = (float*)carve((size_t)NC_ * HD_ * 4);
  float*  ad_cq   = (float*)carve((size_t)NQ_ * HD_ * 4);
  float*  as_qc   = (float*)carve((size_t)NQ_ * HD_ * 4);
  float*  ad_qc   = (float*)carve((size_t)NC_ * HD_ * 4);
  float*  as_cc   = (float*)carve((size_t)NC_ * HD_ * 4);
  float*  ad_cc   = (float*)carve((size_t)NC_ * HD_ * 4);
  unsigned* um_cq = (unsigned*)carve((size_t)NQ_ * HD_ * 4);
  float*  dn_cq   = (float*)carve((size_t)NQ_ * HD_ * 4);
  unsigned* um_qc = (unsigned*)carve((size_t)NC_ * HD_ * 4);
  float*  dn_qc   = (float*)carve((size_t)NC_ * HD_ * 4);
  unsigned* um_cc = (unsigned*)carve((size_t)NC_ * HD_ * 4);
  float*  dn_cc   = (float*)carve((size_t)NC_ * HD_ * 4);
  float*  agg_cq  = (float*)carve((size_t)NQ_ * CH_ * 4);
  float*  agg_qc  = (float*)carve((size_t)NC_ * CH_ * 4);
  float*  agg_cc  = (float*)carve((size_t)NC_ * CH_ * 4);
  float*  cs_c    = (float*)carve(2 * 128 * 4);
  float*  cs_q    = (float*)carve(128 * 4);
  float*  at_c    = (float*)carve(2 * 4);
  float*  at_q    = (float*)carve(1 * 4);

  auto cvt = [&](const float* in, half_t* o, int n){
    cvt16_kernel<<<(n + 255) / 256, 256, 0, stream>>>(in, o, n);
  };
  auto cvtT = [&](const float* in, half_t* o, int N){   // weights [128,N] -> [N,128] f16
    cvtT_kernel<<<(128 * N + 255) / 256, 256, 0, stream>>>(in, o, N);
  };
  auto gemm = [&](const half_t* A, const half_t* Bt, const float* bias, float* o,
                  float* colsum, int M, int N, int mode){
    dim3 blk(32, 4), grd(N / 16, (M / 16 + MT_ * 4 - 1) / (MT_ * 4));
    wmma_gemm128<<<grd, blk, 0, stream>>>(A, Bt, bias, o, colsum, M, N, mode);
  };
  auto attsc = [&](const float* h, const float* att, float* o, int n){
    att_score_kernel<<<(n * HD_ + 255) / 256, 256, 0, stream>>>(h, att, o, n);
  };
  auto emax = [&](const int* ei, const float* as, const float* ad, unsigned* um, int E){
    edge_max_kernel<<<(E * HD_ + 255) / 256, 256, 0, stream>>>(ei, ei + E, as, ad, um, E);
  };
  auto escat = [&](const int* ei, const float* as, const float* ad, const unsigned* um,
                   const float* hs, float* dn, float* ag, int E){
    dim3 blk(32, 8), grd((E + 7) / 8);
    edge_scatter_kernel<<<grd, blk, 0, stream>>>(ei, ei + E, as, ad, um, hs, dn, ag, E);
  };
  auto fin = [&](float* ag, const float* dn, int n){
    finalize_kernel<<<(n * CH_ + 255) / 256, 256, 0, stream>>>(ag, dn, n);
  };

  half_t* in_c16 = xc16;  half_t* in_q16 = xq16;
  cvt(xc, xc16, NC_ * CH_);
  cvt(xq, xq16, NQ_ * CH_);

  const HanP* layers[2] = { &h1, &h2 };
  for (int L = 0; L < 2; ++L){
    const HanP& P = *layers[L];
    // zero accumulators (capture-safe async memsets)
    hipMemsetAsync(um_cq, 0, (size_t)NQ_ * HD_ * 4, stream);
    hipMemsetAsync(dn_cq, 0, (size_t)NQ_ * HD_ * 4, stream);
    hipMemsetAsync(um_qc, 0, (size_t)NC_ * HD_ * 4, stream);
    hipMemsetAsync(dn_qc, 0, (size_t)NC_ * HD_ * 4, stream);
    hipMemsetAsync(um_cc, 0, (size_t)NC_ * HD_ * 4, stream);
    hipMemsetAsync(dn_cc, 0, (size_t)NC_ * HD_ * 4, stream);
    hipMemsetAsync(agg_cq, 0, (size_t)NQ_ * CH_ * 4, stream);
    hipMemsetAsync(agg_qc, 0, (size_t)NC_ * CH_ * 4, stream);
    hipMemsetAsync(agg_cc, 0, (size_t)NC_ * CH_ * 4, stream);
    hipMemsetAsync(cs_c, 0, 2 * 128 * 4, stream);
    hipMemsetAsync(cs_q, 0, 128 * 4, stream);

    // node projections (WMMA, weights pre-transposed to [N,128] f16)
    cvtT(P.proj_c_w, wbufA, CH_);
    gemm(in_c16, wbufA, P.proj_c_b, h_c, nullptr, NC_, CH_, 0);
    cvtT(P.proj_q_w, wbufB, CH_);
    gemm(in_q16, wbufB, P.proj_q_b, h_q, nullptr, NQ_, CH_, 0);

    // per-node attention coefficients
    attsc(h_c, P.att_src_cq, as_cq, NC_);
    attsc(h_q, P.att_dst_cq, ad_cq, NQ_);
    attsc(h_q, P.att_src_qc, as_qc, NQ_);
    attsc(h_c, P.att_dst_qc, ad_qc, NC_);
    attsc(h_c, P.att_src_cc, as_cc, NC_);
    attsc(h_c, P.att_dst_cc, ad_cc, NC_);

    // segment max -> exp/denom/scatter -> normalize+relu
    emax(ei_cq, as_cq, ad_cq, um_cq, ECQ_);
    emax(ei_qc, as_qc, ad_qc, um_qc, EQC_);
    emax(ei_cc, as_cc, ad_cc, um_cc, ECC_);
    escat(ei_cq, as_cq, ad_cq, um_cq, h_c, dn_cq, agg_cq, ECQ_);
    escat(ei_qc, as_qc, ad_qc, um_qc, h_q, dn_qc, agg_qc, EQC_);
    escat(ei_cc, as_cc, ad_cc, um_cc, h_c, dn_cc, agg_cc, ECC_);
    fin(agg_cq, dn_cq, NQ_);
    fin(agg_qc, dn_qc, NC_);
    fin(agg_cc, dn_cc, NC_);

    // semantic attention: tanh(x@kW+kb) column sums via WMMA epilogue (no y materialized)
    cvtT(P.k_w, wbufA, CH_);
    cvt(agg_qc, xc16, NC_ * CH_);                  // reuse: proj inputs consumed above
    gemm(xc16, wbufA, P.k_b, nullptr, cs_c + 0,   NC_, CH_, 1);
    cvt(agg_cc, cc16, NC_ * CH_);
    gemm(cc16, wbufA, P.k_b, nullptr, cs_c + 128, NC_, CH_, 1);
    cvt(agg_cq, xq16, NQ_ * CH_);
    gemm(xq16, wbufA, P.k_b, nullptr, cs_q,       NQ_, CH_, 1);
    semantic_attn_kernel<<<1, 128, 0, stream>>>(cs_c, P.q, at_c, 2, 1.0f / NC_);
    semantic_attn_kernel<<<1, 128, 0, stream>>>(cs_q, P.q, at_q, 1, 1.0f / NQ_);

    // combine (+GELU between layers), emit f16 inputs for next GEMM stage
    int do_gelu = (L == 0) ? 1 : 0;
    combine_kernel<<<(NC_ * CH_ + 255) / 256, 256, 0, stream>>>(agg_qc, agg_cc, at_c, 2,
                                                                NC_ * CH_, xc16, do_gelu);
    combine_kernel<<<(NQ_ * CH_ + 255) / 256, 256, 0, stream>>>(agg_cq, nullptr, at_q, 1,
                                                                NQ_ * CH_, xq16, do_gelu);
    in_c16 = xc16; in_q16 = xq16;
  }

  // final projections [N,128] x [128,768] (WMMA)
  cvtT(cproj_w, wbufA, HID_);
  gemm(in_c16, wbufA, cproj_b, out_c, nullptr, NC_, HID_, 0);
  cvtT(qproj_w, wbufB, HID_);
  gemm(in_q16, wbufB, qproj_b, out_q, nullptr, NQ_, HID_, 0);
}
